// MIKT_32976758898822
// MI455X (gfx1250) — compile-verified, matrix operand order checked by reading, not run on hardware
//
#include <hip/hip_runtime.h>

typedef __attribute__((ext_vector_type(16))) __bf16 v16bf;
typedef __attribute__((ext_vector_type(8)))  float  v8f;

#define Bn   128
#define L1n  201
#define LM1  200
#define Qn   4000
#define Cn   128
#define Dn   128

// ---- dynamic LDS layout for the persistent scan kernel (one WG per batch elem) ----
#define OFF_STATE   0                      // f32  [128][132]  (pad 132 floats = 528B, conflict-free)
#define SZ_STATE    (128*132*4)            // 67584
#define OFF_STATEB  (OFF_STATE + SZ_STATE) // bf16 [128][136]  (pad 136 = 272B, 16B aligned)
#define SZ_BF       (128*136*2)            // 34816
#define OFF_GAPB    (OFF_STATEB + SZ_BF)
#define OFF_HSB     (OFF_GAPB  + SZ_BF)
#define OFF_W1AT    (OFF_HSB   + SZ_BF)    // bf16 [128][264]  (W1^T, K=256 padded to 264)
#define SZ_W1AT     (128*264*2)            // 67584
#define OFF_W2T     (OFF_W1AT + SZ_W1AT)   // bf16 [128][136]
#define OFF_VEC     (OFF_W2T  + SZ_BF)     // 15 x 128 f32 vectors
#define SZ_VEC      (15*128*4)
#define SMEM_TOTAL  (OFF_VEC + SZ_VEC)     // 282112 bytes < 320KB WGP LDS

__device__ __forceinline__ float sigm(float x){ return 1.f/(1.f+__expf(-x)); }

__device__ __forceinline__ __bf16 f2bf(float f){
  union { float f; unsigned u; } a; a.f=f;
  unsigned u=a.u;
  unsigned short h=(unsigned short)((u + 0x7fffu + ((u>>16)&1u))>>16);
  union { unsigned short s; __bf16 b; } o; o.s=h; return o.b;
}

// Load one 16x32 bf16 WMMA fragment (A- or B-operand) from a row-major LDS buffer.
// Lane l (wave32): row/col = l&15; K-base = (l&16)?8:0; elems 0..7 -> K=kb..kb+7,
// elems 8..15 -> K=kb+16..kb+23  (matches CDNA5 16-bit A/B VGPR layout).
__device__ __forceinline__ v16bf ld_frag(const __bf16* base, int stride_elems, int lane){
  const __bf16* p = base + (lane & 15) * stride_elems + ((lane & 16) ? 8 : 0);
  v16bf a;
  uint4* ap = reinterpret_cast<uint4*>(&a);
  ap[0] = *reinterpret_cast<const uint4*>(p);
  ap[1] = *reinterpret_cast<const uint4*>(p + 16);
  return a;
}

// masked softmax over x[0..127] in LDS (works for blockDim 128 or 256)
__device__ __forceinline__ void softmax128(float* x, float* red, int tid){
  if (tid<128) red[tid]=x[tid];
  __syncthreads();
  #pragma unroll
  for (int s=64;s>0;s>>=1){ if (tid<s) red[tid]=fmaxf(red[tid],red[tid+s]); __syncthreads(); }
  float mx = red[0];
  __syncthreads();
  float e = 0.f;
  if (tid<128){ e = __expf(x[tid]-mx); red[tid]=e; }
  __syncthreads();
  #pragma unroll
  for (int s=64;s>0;s>>=1){ if (tid<s) red[tid]+=red[tid+s]; __syncthreads(); }
  float sum = red[0];
  __syncthreads();
  if (tid<128) x[tid]=e/sum;
  __syncthreads();
}

// ------------------------- setup kernels -------------------------

__global__ void k_transpose_f32(const float* __restrict__ src, float* __restrict__ dst,
                                int rows, int cols){
  int i = blockIdx.x*256 + threadIdx.x;
  if (i < rows*cols){ int r=i/cols, c=i-r*cols; dst[c*rows+r]=src[i]; }
}

__global__ void k_transpose_bf16(const float* __restrict__ src, __bf16* __restrict__ dst,
                                 int rows, int cols){
  int i = blockIdx.x*256 + threadIdx.x;
  if (i < rows*cols){ int r=i/cols, c=i-r*cols; dst[c*rows+r]=f2bf(src[i]); }
}

// q_skill = skill_embed @ skill_linear_w + b   (128x128, one-time)
__global__ void k_qskill(const float* __restrict__ skE, const float* __restrict__ slw,
                         const float* __restrict__ slb, float* __restrict__ qskill){
  int idx = blockIdx.x*256 + threadIdx.x;
  if (idx >= Cn*Dn) return;
  int c = idx>>7, d = idx&127;
  float s = slb[d];
  for (int k=0;k<128;k++) s += skE[c*128+k]*slw[k*128+d];
  qskill[idx]=s;
}

// pro_fused[r] = attn_r @ skill_embed + sigmoid(pdw[r]) * (skill_mean_r @ pro_change_w + b)
__global__ void __launch_bounds__(128)
k_pro_fused(const float* __restrict__ Qt, const float* __restrict__ proE,
            const float* __restrict__ skE, const float* __restrict__ pdw,
            const float* __restrict__ qskill, const float* __restrict__ plT,
            const float* __restrict__ plb, const float* __restrict__ pcT,
            const float* __restrict__ pcb, float* __restrict__ fused)
{
  __shared__ float srow[128], pe[128], qp[128], sm[128], attn[128], red[128];
  const int r = blockIdx.x, tid = threadIdx.x;
  srow[tid] = Qt[r*128+tid];
  pe[tid]   = proE[r*128+tid];
  red[tid]  = srow[tid];
  __syncthreads();
  #pragma unroll
  for (int s=64;s>0;s>>=1){ if (tid<s) red[tid]+=red[tid+s]; __syncthreads(); }
  const float ssum = red[0];
  __syncthreads();
  { // skill_mean (column d = tid)
    float s=0.f;
    for (int c=0;c<128;c++) s += srow[c]*skE[c*128+tid];
    sm[tid] = s/(ssum+1e-8f);
  }
  { // q_pro row r, output d = tid
    const float* wr = plT + tid*128;
    float s = plb[tid];
    for (int k=0;k<128;k++) s += pe[k]*wr[k];
    qp[tid]=s;
  }
  __syncthreads();
  { // masked logits over skills c = tid
    float s = -1e9f;
    if (srow[tid] != 0.f){
      s = 0.f;
      const float* qs = qskill + tid*128;
      for (int d=0;d<128;d++) s += qp[d]*qs[d];
      s *= 0.088388347648318447f;   // 1/sqrt(128)
    }
    attn[tid]=s;
  }
  __syncthreads();
  softmax128(attn, red, tid);
  {
    float acc=0.f;
    for (int c=0;c<128;c++) acc += attn[c]*skE[c*128+tid];
    const float* wr = pcT + tid*128;
    float s = pcb[tid];
    for (int k=0;k<128;k++) s += sm[k]*wr[k];
    acc += sigm(pdw[r])*s;
    fused[r*128+tid]=acc;
  }
}

// ------------------------- persistent scan kernel -------------------------
// grid = 128 (one WG per batch element), block = 256 threads = 8 waves.
__global__ void __launch_bounds__(256)
k_scan(const int* __restrict__ qseq, const int* __restrict__ cseq,
       const float* __restrict__ Qt, const float* __restrict__ fused,
       const float* __restrict__ pdw, const float* __restrict__ ansE,
       const float* __restrict__ timeS, const float* __restrict__ s0,
       const float* __restrict__ a0,
       const __bf16* __restrict__ w1aTg, const float* __restrict__ w1cT,
       const __bf16* __restrict__ w2Tg,
       const float* __restrict__ sfb1, const float* __restrict__ sfb2,
       const float* __restrict__ allfT1, const float* __restrict__ afb1,
       const float* __restrict__ allfT2, const float* __restrict__ afb2,
       const float* __restrict__ predT,  const float* __restrict__ pab,
       const float* __restrict__ abilT1, const float* __restrict__ abb1,
       const float* __restrict__ abw2,   const float* __restrict__ abb2,
       const float* __restrict__ obtT,   const float* __restrict__ obb,
       const float* __restrict__ nowT1,  const float* __restrict__ nob1,
       const float* __restrict__ nowT2,  const float* __restrict__ nob2,
       float* __restrict__ outp)
{
  extern __shared__ char smem[];
  float*  state  = (float*)(smem + OFF_STATE);    // fp32 master state [128][132]
  __bf16* stateB = (__bf16*)(smem + OFF_STATEB);  // bf16 shadow (GEMM1 A, K 0..127)
  __bf16* gapB   = (__bf16*)(smem + OFF_GAPB);    // gap embeds   (GEMM1 A, K 128..255)
  __bf16* hsB    = (__bf16*)(smem + OFF_HSB);     // relu output  (GEMM2 A)
  __bf16* w1aT   = (__bf16*)(smem + OFF_W1AT);    // W1[0:256]^T  staged once
  __bf16* w2T    = (__bf16*)(smem + OFF_W2T);     // W2^T         staged once
  float*  vec    = (float*)(smem + OFF_VEC);
  float* allst = vec;        float* fga  = vec+128;  float* nowp = vec+256;
  float* nowX  = vec+384;    float* bias1= vec+512;  float* q2s  = vec+640;
  float* scv   = vec+768;    float* need = vec+896;  float* pav  = vec+1024;
  float* tmp1  = vec+1152;   float* toget= vec+1280; float* agap = vec+1408;
  float* red   = vec+1536;
  int*   lst   = (int*)(vec+1664);
  int*   gidx  = (int*)(vec+1792);

  const int b    = blockIdx.x;
  const int tid  = threadIdx.x;
  const int wave = tid>>5, lane = tid&31;
  const int laneN = lane&15;
  const int rowHi = (lane&16)?8:0;

  // ---- init: state + shadow, staged weights, vectors ----
  for (int i=tid;i<128*128;i+=256){
    int c=i>>7,d=i&127; float v=s0[i];
    state[c*132+d]=v; stateB[c*136+d]=f2bf(v);
  }
  for (int i=tid;i<128*256;i+=256){ int n=i>>8,k=i&255; w1aT[n*264+k]=w1aTg[i]; }
  for (int i=tid;i<128*128;i+=256){ int n=i>>7,k=i&127; w2T[n*136+k]=w2Tg[i]; }
  if (tid<128){ allst[tid]=a0[tid]; lst[tid]=0; agap[tid]=timeS[128+tid]; }
  __syncthreads();

  for (int t=0;t<LM1;t++){
    const int q   = qseq[b*L1n + t + 1];
    const int ans = cseq[b*L1n + t + 1];

    if (tid<128){
      float pv = fused[q*128+tid];
      nowp[tid]=pv;
      nowX[tid]=pv + ansE[ans*128+tid];
      float m = Qt[q*128+tid];
      q2s[tid]=m;
      gidx[tid]= t - ((m!=0.f)? lst[tid] : 0);
    }
    __syncthreads();

    // gap embeddings -> bf16 (rows c = skills)
    for (int i=tid;i<128*128;i+=256){
      int c=i>>7,k=i&127;
      gapB[c*136+k]=f2bf(timeS[gidx[c]*128+k]);
    }

    // ---- small matvec tail: forget_all and folded bias ----
    if (tid<128){
      const float* wr = allfT1 + tid*256;
      float s = afb1[tid];
      for (int k=0;k<128;k++) s += allst[k]*wr[k];
      for (int k=0;k<128;k++) s += agap[k]*wr[128+k];
      tmp1[tid] = fmaxf(s,0.f);
    }
    __syncthreads();
    if (tid<128){
      const float* wr = allfT2 + tid*128;
      float s = afb2[tid];
      for (int k=0;k<128;k++) s += tmp1[k]*wr[k];
      fga[tid] = allst[tid]*sigm(s);
    }
    __syncthreads();
    if (tid<128){ // bias1 = skill_forget_b1 + forget_all @ W1[256:384]  (rank-1 fold)
      const float* wr = w1cT + tid*128;
      float s = sfb1[tid];
      for (int k=0;k<128;k++) s += fga[k]*wr[k];
      bias1[tid]=s;
    }
    __syncthreads();

    // ---- GEMM1: hs = relu([state|gap] @ W1a + bias1)  M=128 K=256 N=128 ----
    {
      v8f acc[8] = {};
      #pragma unroll
      for (int kt=0;kt<8;kt++){
        const __bf16* ab = (kt<4) ? (stateB + wave*16*136 + kt*32)
                                  : (gapB   + wave*16*136 + (kt-4)*32);
        v16bf af = ld_frag(ab, 136, lane);
        #pragma unroll
        for (int nt=0;nt<8;nt++){
          v16bf bf = ld_frag(w1aT + nt*16*264 + kt*32, 264, lane);
          acc[nt] = __builtin_amdgcn_wmma_f32_16x16x32_bf16(
                      false, af, false, bf, (short)0, acc[nt], false, false);
        }
      }
      #pragma unroll
      for (int nt=0;nt<8;nt++){
        float bc = bias1[nt*16+laneN];
        #pragma unroll
        for (int e=0;e<8;e++){
          int row = wave*16 + rowHi + e;
          hsB[row*136 + nt*16 + laneN] = f2bf(fmaxf(acc[nt][e] + bc, 0.f));
        }
      }
    }
    __syncthreads();

    // ---- GEMM2: sf = sigmoid(hs @ W2 + b2); state *= masked sf ----
    {
      v8f acc[8] = {};
      #pragma unroll
      for (int kt=0;kt<4;kt++){
        v16bf af = ld_frag(hsB + wave*16*136 + kt*32, 136, lane);
        #pragma unroll
        for (int nt=0;nt<8;nt++){
          v16bf bf = ld_frag(w2T + nt*16*136 + kt*32, 136, lane);
          acc[nt] = __builtin_amdgcn_wmma_f32_16x16x32_bf16(
                      false, af, false, bf, (short)0, acc[nt], false, false);
        }
      }
      #pragma unroll
      for (int nt=0;nt<8;nt++){
        float b2 = sfb2[nt*16+laneN];
        #pragma unroll
        for (int e=0;e<8;e++){
          int row = wave*16 + rowHi + e;
          float sf = sigm(acc[nt][e] + b2);
          if (q2s[row]==0.f) sf = 1.f;
          state[row*132 + nt*16 + laneN] *= sf;
        }
      }
    }
    __syncthreads();

    // ---- attention sc -> need ----
    if (tid<128){
      float m=q2s[tid]; float s=-1e9f;
      if (m!=0.f){
        s=0.f; const float* sr = state + tid*132;
        for (int d=0;d<128;d++) s += nowp[d]*sr[d];
        s *= (1.f/128.f);
      }
      scv[tid]=s;
    }
    __syncthreads();
    softmax128(scv, red, tid);
    if (tid<128){
      float s=0.f;
      for (int c=0;c<128;c++) s += scv[c]*state[c*132+tid];
      need[tid]=s;
    }
    __syncthreads();

    // ---- pa, ha, ability, output ----
    if (tid<128){
      const float* wr = predT + tid*384;
      float s = pab[tid];
      for (int k=0;k<128;k++) s += need[k]*wr[k];
      for (int k=0;k<128;k++) s += fga[k]*wr[128+k];
      for (int k=0;k<128;k++) s += nowp[k]*wr[256+k];
      pav[tid]=sigm(s);
    }
    __syncthreads();
    if (tid<128){
      const float* wr = abilT1 + tid*384;
      float s = abb1[tid];
      for (int k=0;k<128;k++) s += (1.f-pav[k])*need[k]*wr[k];
      for (int k=0;k<128;k++) s += pav[k]*fga[k]*wr[128+k];
      for (int k=0;k<128;k++) s += nowp[k]*wr[256+k];
      tmp1[tid]=fmaxf(s,0.f);
    }
    __syncthreads();
    {
      float v = (tid<128)? tmp1[tid]*abw2[tid] : 0.f;
      if (tid<128) red[tid]=v;
      __syncthreads();
      #pragma unroll
      for (int s=64;s>0;s>>=1){ if (tid<s) red[tid]+=red[tid+s]; __syncthreads(); }
      if (tid==0){
        float ability = sigm(red[0] + abb2[0]);
        float diff    = sigm(pdw[q]);
        outp[b*LM1 + t] = sigm(5.f*(ability-diff));
      }
      __syncthreads();
    }

    // ---- state carries for next step ----
    if (tid<128 && q2s[tid]!=0.f) lst[tid]=t;
    if (tid<128){
      const float* wr = obtT + tid*128;
      float s = obb[tid];
      for (int k=0;k<128;k++) s += nowX[k]*wr[k];
      allst[tid] = fga[tid] + tanhf(s);
      const float* w1 = nowT1 + tid*128;
      float s1 = nob1[tid];
      for (int k=0;k<128;k++) s1 += nowX[k]*w1[k];
      tmp1[tid]=tanhf(s1);
    }
    __syncthreads();
    if (tid<128){
      const float* wr = nowT2 + tid*128;
      float s = nob2[tid];
      for (int k=0;k<128;k++) s += tmp1[k]*wr[k];
      toget[tid]=tanhf(s);
    }
    __syncthreads();
    if (tid<128){
      float m=q2s[tid]; float s=-1e9f;
      if (m!=0.f){
        s=0.f; const float* sr = state + tid*132;
        for (int d=0;d<128;d++) s += toget[d]*sr[d];
        s *= (1.f/128.f);
      }
      scv[tid]=s;
    }
    __syncthreads();
    softmax128(scv, red, tid);
    // state += sc2 (x) to_get ; refresh bf16 shadow for next step's GEMM1
    for (int i=tid;i<128*128;i+=256){
      int c=i>>7,d=i&127;
      float nv = state[c*132+d] + scv[c]*toget[d];
      state[c*132+d]=nv;
      stateB[c*136+d]=f2bf(nv);
    }
    __syncthreads();
  }
}

// ------------------------- host launcher -------------------------

extern "C" void kernel_launch(void* const* d_in, const int* in_sizes, int n_in,
                              void* d_out, int out_size, void* d_ws, size_t ws_size,
                              hipStream_t stream)
{
  (void)in_sizes; (void)n_in; (void)out_size; (void)ws_size;

  const int*   qseq  = (const int*)  d_in[0];
  const int*   cseq  = (const int*)  d_in[1];
  const float* Qt    = (const float*)d_in[2];
  const float* proE  = (const float*)d_in[3];
  const float* skE   = (const float*)d_in[4];
  const float* pdw   = (const float*)d_in[5];
  const float* ansE  = (const float*)d_in[6];
  const float* timeS = (const float*)d_in[7];
  const float* s0    = (const float*)d_in[8];
  const float* a0    = (const float*)d_in[9];
  const float* plw   = (const float*)d_in[10]; const float* plb  = (const float*)d_in[11];
  const float* slw   = (const float*)d_in[12]; const float* slb  = (const float*)d_in[13];
  const float* pcw   = (const float*)d_in[14]; const float* pcb  = (const float*)d_in[15];
  const float* afw1  = (const float*)d_in[16]; const float* afb1 = (const float*)d_in[17];
  const float* afw2  = (const float*)d_in[18]; const float* afb2 = (const float*)d_in[19];
  const float* sfw1  = (const float*)d_in[20]; const float* sfb1 = (const float*)d_in[21];
  const float* sfw2  = (const float*)d_in[22]; const float* sfb2 = (const float*)d_in[23];
  const float* paw   = (const float*)d_in[24]; const float* pabv = (const float*)d_in[25];
  const float* abw1  = (const float*)d_in[26]; const float* abb1 = (const float*)d_in[27];
  const float* abw2  = (const float*)d_in[28]; const float* abb2 = (const float*)d_in[29];
  const float* obw   = (const float*)d_in[30]; const float* obb  = (const float*)d_in[31];
  const float* now1  = (const float*)d_in[32]; const float* nob1 = (const float*)d_in[33];
  const float* now2  = (const float*)d_in[34]; const float* nob2 = (const float*)d_in[35];

  char* ws = (char*)d_ws;
  size_t off = 0;
  auto alloc = [&](size_t bytes)->char*{
    char* p = ws + off; off = (off + bytes + 255) & ~size_t(255); return p;
  };
  float*  fused  = (float*) alloc((size_t)Qn*Dn*4);
  float*  qskill = (float*) alloc(128*128*4);
  float*  plT    = (float*) alloc(128*128*4);
  float*  pcT    = (float*) alloc(128*128*4);
  float*  allfT1 = (float*) alloc(128*256*4);
  float*  allfT2 = (float*) alloc(128*128*4);
  float*  w1cT   = (float*) alloc(128*128*4);
  float*  predT  = (float*) alloc(128*384*4);
  float*  abilT1 = (float*) alloc(128*384*4);
  float*  obtT   = (float*) alloc(128*128*4);
  float*  nowT1  = (float*) alloc(128*128*4);
  float*  nowT2  = (float*) alloc(128*128*4);
  __bf16* w1aT   = (__bf16*)alloc(128*256*2);
  __bf16* w2T    = (__bf16*)alloc(128*128*2);

  auto tr = [&](const float* s, float* d, int r, int c){
    k_transpose_f32<<<(r*c+255)/256, 256, 0, stream>>>(s, d, r, c);
  };
  tr(plw,  plT,    128,128);
  tr(pcw,  pcT,    128,128);
  tr(afw1, allfT1, 256,128);
  tr(afw2, allfT2, 128,128);
  tr(sfw1 + 256*128, w1cT, 128,128);   // W1[256:384] (eff rows, folded into bias)
  tr(paw,  predT,  384,128);
  tr(abw1, abilT1, 384,128);
  tr(obw,  obtT,   128,128);
  tr(now1, nowT1,  128,128);
  tr(now2, nowT2,  128,128);
  k_transpose_bf16<<<(256*128+255)/256, 256, 0, stream>>>(sfw1, w1aT, 256, 128);
  k_transpose_bf16<<<(128*128+255)/256, 256, 0, stream>>>(sfw2, w2T,  128, 128);

  k_qskill<<<64, 256, 0, stream>>>(skE, slw, slb, qskill);
  k_pro_fused<<<Qn, 128, 0, stream>>>(Qt, proE, skE, pdw, qskill, plT, plb, pcT, pcb, fused);

  k_scan<<<Bn, 256, SMEM_TOTAL, stream>>>(
      qseq, cseq, Qt, fused, pdw, ansE, timeS, s0, a0,
      w1aT, w1cT, w2T, sfb1, sfb2,
      allfT1, afb1, allfT2, afb2,
      predT, pabv, abilT1, abb1, abw2, abb2,
      obtT, obb, nowT1, nob1, nowT2, nob2,
      (float*)d_out);
}
